// MultiHeadAttention_88708254532086
// MI455X (gfx1250) — matrix operand order repaired and checked by reading, not compiled, not run on hardware
//
#include <hip/hip_runtime.h>

// ---------------------------------------------------------------------------
// MHA forward for MI455X (gfx1250, wave32, WMMA + Tensor Data Mover).
//   x:[2,2048,1024] f32, Wq/Wk/Wv:[1024,1024] f32 -> out:[2,2048,1024] f32
// bf16 WMMA (v_wmma_f32_16x16x32_bf16) for all GEMMs; flash-attention online
// softmax (no SxS matrix in HBM); K/V tiles staged global->LDS with
// tensor_load_to_lds (TDM, TENSORcnt) double-buffered against the WMMA work.
// Workspace: Q,K as bf16 [B][H][S][Dh]; V stored transposed [B][H][Dh][S].
// ---------------------------------------------------------------------------

typedef __bf16 bf16_t;
typedef __attribute__((ext_vector_type(16))) __bf16 v16bf;
typedef __attribute__((ext_vector_type(8)))  __bf16 v8bf;
typedef __attribute__((ext_vector_type(4)))  __bf16 v4bf;
typedef __attribute__((ext_vector_type(8)))  float  v8f;
typedef __attribute__((ext_vector_type(4)))  unsigned v4u;
typedef __attribute__((ext_vector_type(8)))  unsigned v8u;

#define BATCH   2
#define SEQ     2048
#define DIN     1024
#define DOUT    1024
#define NHEAD   16
#define HDIM    64

static __device__ __forceinline__ bf16_t f2bf(float f) {
  unsigned u = __builtin_bit_cast(unsigned, f);
  u += 0x7fffu + ((u >> 16) & 1u);          // round-to-nearest-even
  unsigned short h = (unsigned short)(u >> 16);
  return __builtin_bit_cast(bf16_t, h);
}

static __device__ __forceinline__ v16bf combine16(v8bf lo, v8bf hi) {
  v16bf a;
#pragma unroll
  for (int i = 0; i < 8; ++i) { a[i] = lo[i]; a[i + 8] = hi[i]; }
  return a;
}

// A-fragment (16x32 bf16): lane%16 = row M, lane-half selects K 16..31 (k = 16*half + s).
static __device__ __forceinline__ v16bf load_afrag(const bf16_t* row, int half) {
  v8bf lo = *(const v8bf*)(row + half * 16);
  v8bf hi = *(const v8bf*)(row + half * 16 + 8);
  return combine16(lo, hi);
}

// B-fragment (32x16 bf16) from an [N][K]-major tile; `p` includes half*16.
static __device__ __forceinline__ v16bf load_bfrag(const bf16_t* p) {
  v8bf lo = *(const v8bf*)(p);
  v8bf hi = *(const v8bf*)(p + 8);
  return combine16(lo, hi);
}

static __device__ __forceinline__ v8f wmma_bf16(v16bf a, v16bf b, v8f c) {
  return __builtin_amdgcn_wmma_f32_16x16x32_bf16(
      false, a, false, b, (short)0, c, false, false);
}

// Generic pointer into __shared__ -> wave-relative LDS byte offset
// (LDS aperture keeps the offset in addr[31:0]).
static __device__ __forceinline__ unsigned lds_byte_off(const void* p) {
  return (unsigned)(unsigned long long)p;
}

// ---------------------------------------------------------------------------
// TDM: async 2D tile copy global -> LDS (bf16 elements, row length 64 elts).
// D# per CDNA5 ISA. LDS padding +4 DWORDs every
// 32 DWORDs reproduces the stride-72 (144 B) padded rows the fragment
// loaders use. Groups 2/3 omitted (2D tensor). Tracked by TENSORcnt.
// ---------------------------------------------------------------------------
static __device__ __forceinline__ void tdm_load_2d_64(
    unsigned lds_off, const void* gaddr,
    unsigned tensor_d0, unsigned tensor_d1, unsigned long long stride0) {
  unsigned long long ga = (unsigned long long)gaddr;
  v4u g0;
  g0[0] = 1u;                                     // count=1 (valid user D#)
  g0[1] = lds_off;                                // lds_addr
  g0[2] = (unsigned)ga;                           // global_addr[31:0]
  g0[3] = (unsigned)(ga >> 32) | 0x80000000u;     // addr[56:32] | type=2
  v8u g1;
  g1[0] = (1u << 16)      // data_size = 2 bytes
        | (1u << 20)      // pad_enable
        | (4u << 22)      // pad_interval: 32 DWORDs (=64 bf16 = one row)
        | (3u << 25);     // pad_amount: 4 DWORDs (=8 bf16)
  g1[1] = (tensor_d0 & 0xffffu) << 16;            // tensor_dim0 lo16
  g1[2] = (tensor_d0 >> 16) | ((tensor_d1 & 0xffffu) << 16);
  g1[3] = (tensor_d1 >> 16) | (64u << 16);        // tile_dim0 = 64
  g1[4] = 64u;                                    // tile_dim1 = 64, tile_dim2=0
  g1[5] = (unsigned)stride0;                      // tensor_dim0_stride lo32
  g1[6] = (unsigned)(stride0 >> 32) & 0xffffu;    // stride hi16, dim1_stride=0
  g1[7] = 0u;
  asm volatile("tensor_load_to_lds %0, %1" :: "s"(g0), "s"(g1) : "memory");
}

// ---------------------------------------------------------------------------
// Kernel 1: QKV projection. grid = (N/64, M/128, 3), block = 256 (8 waves).
// z==0 -> Q (scaled 1/sqrt(Dh), [B][H][S][Dh]); z==1 -> K ([B][H][S][Dh]);
// z==2 -> V stored TRANSPOSED [B][H][Dh][S] so the attention kernel's PV
// B-tiles are contiguous row copies (TDM-friendly, no LDS transpose).
// ---------------------------------------------------------------------------
__global__ __launch_bounds__(256, 2)
void qkv_proj_wmma(const float* __restrict__ x,
                   const float* __restrict__ Wq,
                   const float* __restrict__ Wk,
                   const float* __restrict__ Wv,
                   bf16_t* __restrict__ Qw,
                   bf16_t* __restrict__ Kw,
                   bf16_t* __restrict__ Vw) {
  __shared__ __align__(16) bf16_t xs[128 * 40];   // [row][k], stride 40
  __shared__ __align__(16) bf16_t wt[64 * 40];    // W transposed: [n][k]

  const int tid  = threadIdx.x;
  const int lane = tid & 31;
  const int wv   = tid >> 5;
  const int half = lane >> 4;
  const int m16  = lane & 15;

  const int n0 = blockIdx.x * 64;
  const int m0 = blockIdx.y * 128;
  const int z  = blockIdx.z;

  const float* W     = (z == 0) ? Wq : ((z == 1) ? Wk : Wv);
  bf16_t*      outp  = (z == 0) ? Qw : ((z == 1) ? Kw : Vw);
  const float  scale = (z == 0) ? 0.125f : 1.0f;

  const v8f vzero = {0.f, 0.f, 0.f, 0.f, 0.f, 0.f, 0.f, 0.f};
  v8f acc[4];
#pragma unroll
  for (int nt = 0; nt < 4; ++nt) acc[nt] = vzero;

  for (int kt = 0; kt < DIN / 32; ++kt) {
    const int k0 = kt * 32;

    // stage x tile [128 x 32] f32 -> bf16
#pragma unroll
    for (int j = 0; j < 4; ++j) {
      const int idx = j * 256 + tid;
      const int row = idx >> 3;
      const int c4  = (idx & 7) * 4;
      const float4 v = *(const float4*)(x + (size_t)(m0 + row) * DIN + k0 + c4);
      v4bf p;
      p[0] = f2bf(v.x); p[1] = f2bf(v.y); p[2] = f2bf(v.z); p[3] = f2bf(v.w);
      *(v4bf*)(xs + row * 40 + c4) = p;
    }
    // stage W tile [32 x 64] transposed -> wt[n][k]
    {
      const int n  = tid & 63;
      const int kb = (tid >> 6) * 8;
      v8bf p;
#pragma unroll
      for (int i = 0; i < 8; ++i)
        p[i] = f2bf(W[(size_t)(k0 + kb + i) * DOUT + n0 + n]);
      *(v8bf*)(wt + n * 40 + kb) = p;
    }
    if (kt + 1 < DIN / 32)                          // L2 warm-up for next chunk
      __builtin_prefetch(x + (size_t)(m0 + (tid >> 1)) * DIN + k0 + 32 +
                         (tid & 1) * 16, 0, 1);
    __syncthreads();

    const v16bf a = load_afrag(xs + (wv * 16 + m16) * 40, half);
#pragma unroll
    for (int nt = 0; nt < 4; ++nt) {
      const v16bf b = load_bfrag(wt + (nt * 16 + m16) * 40 + half * 16);
      acc[nt] = wmma_bf16(a, b, acc[nt]);
    }
    __syncthreads();
  }

  // epilogue: scale, convert, scatter
#pragma unroll
  for (int nt = 0; nt < 4; ++nt) {
#pragma unroll
    for (int r = 0; r < 8; ++r) {
      const int mrow = m0 + wv * 16 + r + 8 * half;   // C: VGPR idx = row
      const int ncol = n0 + nt * 16 + m16;            // C: lane%16 = col
      const int b = mrow >> 11, s = mrow & (SEQ - 1);
      const int head = ncol >> 6, d = ncol & (HDIM - 1);
      size_t oidx;
      if (z == 2)   // V transposed: [B][H][Dh][S]
        oidx = (((size_t)b * NHEAD + head) * HDIM + d) * SEQ + s;
      else          // Q/K: [B][H][S][Dh]
        oidx = (((size_t)b * NHEAD + head) * SEQ + s) * HDIM + d;
      outp[oidx] = f2bf(acc[nt][r] * scale);
    }
  }
}

// ---------------------------------------------------------------------------
// Kernel 2: causal flash attention. grid = (S/128, H, B), block = 256.
// Wave w owns query rows [128*qt + 16w, +16). 64-key KV tiles, K/V staged by
// TDM into double-buffered LDS; wave 0 drives the DMA pipeline.
// ---------------------------------------------------------------------------
__global__ __launch_bounds__(256, 2)
void attn_flash_wmma(const bf16_t* __restrict__ Qw,
                     const bf16_t* __restrict__ Kw,
                     const bf16_t* __restrict__ Vw,
                     float* __restrict__ out) {
  // stride-72 padded rows (TDM pad: +8 bf16 per 64) -> 4608 elts per buffer
  __shared__ __align__(16) bf16_t lk[2 * 64 * 72];   // [key][d]  (B for QK^T)
  __shared__ __align__(16) bf16_t lv[2 * 64 * 72];   // [d][key]  (B for PV)
  __shared__ __align__(16) bf16_t lp[8 * 16 * 72];   // per-wave P staging

  const int tid  = threadIdx.x;
  const int lane = tid & 31;
  const int wv   = tid >> 5;
  const int half = lane >> 4;
  const int m16  = lane & 15;

  const int qt = blockIdx.x;
  const int h  = blockIdx.y;
  const int b  = blockIdx.z;
  const int q0 = qt * 128;

  const bf16_t* Qb  = Qw + (((size_t)b * NHEAD + h) * SEQ) * HDIM;
  const bf16_t* Kb  = Kw + (((size_t)b * NHEAD + h) * SEQ) * HDIM;
  const bf16_t* Vtb = Vw + (((size_t)b * NHEAD + h) * HDIM) * SEQ; // [d][s]

  // Q A-fragments straight from global (rows contiguous bf16).
  const bf16_t* qrow = Qb + (size_t)(q0 + wv * 16 + m16) * HDIM;
  const v16bf qa0 = load_afrag(qrow, half);
  const v16bf qa1 = load_afrag(qrow + 32, half);

  const float NEG_INF = -__builtin_inff();
  const v8f vzero = {0.f, 0.f, 0.f, 0.f, 0.f, 0.f, 0.f, 0.f};
  v8f oacc[4];
#pragma unroll
  for (int nt = 0; nt < 4; ++nt) oacc[nt] = vzero;
  float mrow[8], lrow[8];
#pragma unroll
  for (int r = 0; r < 8; ++r) { mrow[r] = NEG_INF; lrow[r] = 0.f; }

  const int nkv = 2 * qt + 2;                      // causal KV-tile bound

  // prologue: kick off tile 0 into buffer 0
  if (wv == 0) {
    tdm_load_2d_64(lds_byte_off(lk), Kb, 64, SEQ, 64);
    tdm_load_2d_64(lds_byte_off(lv), Vtb, SEQ, HDIM, SEQ);
  }

  for (int t = 0; t < nkv; ++t) {
    const int kv0 = t * 64;
    const int buf = t & 1;
    const bf16_t* lkb = lk + buf * 4608;
    const bf16_t* lvb = lv + buf * 4608;

    // issue next tile into the other buffer, then wait for current tile.
    // TDM ops complete in order per wave: tensorcnt<=2 == "all but the two
    // just-issued are done".
    if (wv == 0) {
      if (t + 1 < nkv) {
        const int nb = buf ^ 1;
        const int kn = kv0 + 64;
        tdm_load_2d_64(lds_byte_off(lk + nb * 4608),
                       Kb + (size_t)kn * HDIM, 64, SEQ, 64);
        tdm_load_2d_64(lds_byte_off(lv + nb * 4608),
                       Vtb + kn, SEQ, HDIM, SEQ);
        __builtin_amdgcn_s_wait_tensorcnt(2);
      } else {
        __builtin_amdgcn_s_wait_tensorcnt(0);
      }
    }
    __syncthreads();                               // tile t visible to all

    // ---- scores S = Q K^T : 8 WMMAs ----
    v8f sacc[4];
#pragma unroll
    for (int nt = 0; nt < 4; ++nt) {
      sacc[nt] = vzero;
      const bf16_t* kp = lkb + (nt * 16 + m16) * 72 + half * 16;
      sacc[nt] = wmma_bf16(qa0, load_bfrag(kp), sacc[nt]);
      sacc[nt] = wmma_bf16(qa1, load_bfrag(kp + 32), sacc[nt]);
    }

    // ---- causal mask + row max (C layout: VGPR r = row, lane%16 = col) ----
    const int qg = q0 + wv * 16 + 8 * half;
    float rmax[8];
#pragma unroll
    for (int r = 0; r < 8; ++r) rmax[r] = NEG_INF;
#pragma unroll
    for (int nt = 0; nt < 4; ++nt) {
      const int kg = kv0 + nt * 16 + m16;
#pragma unroll
      for (int r = 0; r < 8; ++r) {
        float sv = sacc[nt][r];
        if (kg > qg + r) sv = NEG_INF;
        sacc[nt][r] = sv;
        rmax[r] = fmaxf(rmax[r], sv);
      }
    }
#pragma unroll
    for (int r = 0; r < 8; ++r) {
#pragma unroll
      for (int off = 1; off < 16; off <<= 1)
        rmax[r] = fmaxf(rmax[r], __shfl_xor(rmax[r], off, 32));
    }

    // ---- online softmax ----
    float alpha[8], rsum[8];
#pragma unroll
    for (int r = 0; r < 8; ++r) {
      const float mn = fmaxf(mrow[r], rmax[r]);
      alpha[r] = __expf(mrow[r] - mn);
      mrow[r] = mn;
      rsum[r] = 0.f;
    }
#pragma unroll
    for (int nt = 0; nt < 4; ++nt) {
#pragma unroll
      for (int r = 0; r < 8; ++r) {
        const float p = __expf(sacc[nt][r] - mrow[r]);
        sacc[nt][r] = p;
        rsum[r] += p;
      }
    }
#pragma unroll
    for (int r = 0; r < 8; ++r) {
#pragma unroll
      for (int off = 1; off < 16; off <<= 1)
        rsum[r] += __shfl_xor(rsum[r], off, 32);
      lrow[r] = lrow[r] * alpha[r] + rsum[r];
    }
#pragma unroll
    for (int nt = 0; nt < 4; ++nt)
#pragma unroll
      for (int r = 0; r < 8; ++r) oacc[nt][r] *= alpha[r];

    // ---- P: C-layout -> A-layout via per-wave LDS round trip ----
    bf16_t* lpb = lp + wv * 16 * 72;
#pragma unroll
    for (int nt = 0; nt < 4; ++nt)
#pragma unroll
      for (int r = 0; r < 8; ++r)
        lpb[(r + 8 * half) * 72 + nt * 16 + m16] = f2bf(sacc[nt][r]);

    const bf16_t* prow = lpb + m16 * 72;
    const v16bf pa0 = load_afrag(prow, half);
    const v16bf pa1 = load_afrag(prow + 32, half);

    // ---- O += P V : 8 WMMAs ----
#pragma unroll
    for (int nt = 0; nt < 4; ++nt) {
      const bf16_t* vp = lvb + (nt * 16 + m16) * 72 + half * 16;
      oacc[nt] = wmma_bf16(pa0, load_bfrag(vp), oacc[nt]);
      oacc[nt] = wmma_bf16(pa1, load_bfrag(vp + 32), oacc[nt]);
    }
    __syncthreads();   // everyone done with buf before TDM overwrites it
  }

  // ---- epilogue: O / l -> f32 [B][S][H*Dh] ----
#pragma unroll
  for (int nt = 0; nt < 4; ++nt) {
#pragma unroll
    for (int r = 0; r < 8; ++r) {
      const int qg = q0 + wv * 16 + r + 8 * half;
      const int d  = nt * 16 + m16;
      out[((size_t)b * SEQ + qg) * DOUT + h * HDIM + d] =
          oacc[nt][r] / lrow[r];
    }
  }
}

// ---------------------------------------------------------------------------
extern "C" void kernel_launch(void* const* d_in, const int* in_sizes, int n_in,
                              void* d_out, int out_size, void* d_ws, size_t ws_size,
                              hipStream_t stream) {
  (void)in_sizes; (void)n_in; (void)out_size; (void)ws_size;
  const float* x  = (const float*)d_in[0];
  const float* Wq = (const float*)d_in[1];
  const float* Wk = (const float*)d_in[2];
  const float* Wv = (const float*)d_in[3];
  float* out = (float*)d_out;

  bf16_t* ws = (bf16_t*)d_ws;
  const size_t per = (size_t)BATCH * NHEAD * SEQ * HDIM;
  bf16_t* Qw = ws;            // [B][H][S][Dh]
  bf16_t* Kw = ws + per;      // [B][H][S][Dh]
  bf16_t* Vw = ws + 2 * per;  // [B][H][Dh][S] (transposed)

  dim3 g1(DOUT / 64, (BATCH * SEQ) / 128, 3);
  qkv_proj_wmma<<<g1, 256, 0, stream>>>(x, Wq, Wk, Wv, Qw, Kw, Vw);

  dim3 g2(SEQ / 128, NHEAD, BATCH);
  attn_flash_wmma<<<g2, 256, 0, stream>>>(Qw, Kw, Vw, out);
}
